// BDH_27805618274992
// MI455X (gfx1250) — matrix operand order repaired and checked by reading, not compiled, not run on hardware
//
#include <hip/hip_runtime.h>
#include <hip/hip_bf16.h>
#include <math.h>

// ---------------------------------------------------------------------------
// BDH forward for MI455X (gfx1250, wave32, WMMA).
// All large GEMMs run through V_WMMA_F32_16X16X32_BF16 (bf16 in, fp32 accum).
// Weights are converted+transposed to bf16 ONCE per launch so every B-side
// fragment load is two contiguous 16B loads (no strided fp32 + cvt chains in
// the inner loops). Problem is compute bound: ~490 GFLOP vs ~0.5 GB traffic.
// ---------------------------------------------------------------------------

typedef __bf16 bhalf;
typedef __attribute__((ext_vector_type(16))) __bf16 v16bf;
typedef __attribute__((ext_vector_type(8)))  __bf16 v8bf;
typedef __attribute__((ext_vector_type(8)))  float  v8f;
typedef __attribute__((ext_vector_type(4)))  unsigned int u32x4;
typedef __attribute__((ext_vector_type(8)))  int i32x8;
typedef __attribute__((ext_vector_type(4)))  int i32x4;

#define T_SEQ   1024
#define D_EMB   256
#define NHEAD   4
#define N_LAT   8192
#define HN      (NHEAD * N_LAT)     // 32768
#define VOCAB   256
#define N_LAYER 4
#define LN_EPS  1e-5f
#define INV_2PI 0.15915494309189535f
#define TWO_PI  6.28318530717958647f

// ---------------- WMMA fragment loaders (CDNA5 wave32 layouts) --------------

// A 16x32 bf16 (MxK): lanes 0-15 M=lane; VGPR0-3 hold K = kh*8 + 0..7,
// VGPR4-7 hold K = 16 + kh*8 + 0..7  (kh = lane>>4).
__device__ __forceinline__ v16bf load_a_frag(const bhalf* __restrict__ A,
                                             int lda, int row0, int k0, int lane) {
    const int m  = row0 + (lane & 15);
    const int kh = lane >> 4;
    const bhalf* p = A + (size_t)m * lda + (size_t)k0 + kh * 8;
    v8bf lo = *reinterpret_cast<const v8bf*>(p);
    v8bf hi = *reinterpret_cast<const v8bf*>(p + 16);
    v16bf r;
#pragma unroll
    for (int e = 0; e < 8; ++e) { r[e] = lo[e]; r[e + 8] = hi[e]; }
    return r;
}

// B 32x16 (KxN) from a bf16 matrix stored TRANSPOSED (Bt[n, k] row-major):
// lane: N = lane&15, K = (lane>>4)*8 + e and 16 + (lane>>4)*8 + e -> 2x16B loads.
__device__ __forceinline__ v16bf load_bt_frag(const bhalf* __restrict__ Bt,
                                              int ldbt, int k0, int n0, int lane) {
    const int n  = n0 + (lane & 15);
    const int kh = lane >> 4;
    const bhalf* p = Bt + (size_t)n * ldbt + (size_t)k0 + kh * 8;
    v8bf lo = *reinterpret_cast<const v8bf*>(p);
    v8bf hi = *reinterpret_cast<const v8bf*>(p + 16);
    v16bf r;
#pragma unroll
    for (int e = 0; e < 8; ++e) { r[e] = lo[e]; r[e + 8] = hi[e]; }
    return r;
}

// ---------------- GEMM kernel ----------------------------------------------
// Block = 256 threads = 8 waves arranged 2(M) x 4(N); each wave owns a
// 32x64 strip: 2 A-frags x 4 B-frags -> 8 WMMA per 32-deep K step.
// Grid: (Ncols/256, M/64, batch).
// EPI: 0 = store f32, 1 = relu->f32, 2 = strict-lower mask->bf16,
//      3 = relu * gate -> bf16   (gate = x_sp, for xy = x_sp * y_sp)

template <int EPI>
__global__ __launch_bounds__(256) void gemm_wmma(
    const bhalf* __restrict__ A,  int lda,  long long aStr,
    const bhalf* __restrict__ Bt, int ldbt, long long bStr,
    void*        __restrict__ C,  int ldc,  long long cStr,
    const float* __restrict__ gate, int ldg, long long gStr,
    int K) {

    const int lane = threadIdx.x & 31;
    const int wave = threadIdx.x >> 5;
    const int row0 = (blockIdx.y * 2 + (wave >> 2)) * 32;
    const int col0 = blockIdx.x * 256 + (wave & 3) * 64;
    const int z    = blockIdx.z;

    const bhalf* Az = A  + (size_t)z * aStr;
    const bhalf* Bz = Bt + (size_t)z * bStr;

    // Causal-mask fast path: whole 32x64 strip above the diagonal -> zeros.
    if (EPI == 2 && col0 >= row0 + 32) {
        bhalf* Cb = (bhalf*)C + (size_t)z * cStr;
#pragma unroll
        for (int mi = 0; mi < 2; ++mi)
#pragma unroll
            for (int j = 0; j < 4; ++j) {
                const int cn = col0 + j * 16 + (lane & 15);
#pragma unroll
                for (int r = 0; r < 8; ++r) {
                    const int rn = row0 + mi * 16 + r + (lane >> 4) * 8;
                    Cb[(size_t)rn * ldc + cn] = (bhalf)0.0f;
                }
            }
        return;
    }

    v8f acc[2][4];
#pragma unroll
    for (int mi = 0; mi < 2; ++mi)
#pragma unroll
        for (int j = 0; j < 4; ++j)
#pragma unroll
            for (int e = 0; e < 8; ++e) acc[mi][j][e] = 0.0f;

    for (int k0 = 0; k0 < K; k0 += 32) {
        if (k0 + 32 < K) {  // global_prefetch_b8 of next slabs
            __builtin_prefetch(Az + (size_t)(row0 + (lane & 15)) * lda + k0 + 32, 0, 0);
            __builtin_prefetch(Bz + (size_t)(col0 + (lane & 15)) * ldbt + k0 + 32, 0, 0);
        }
        v16bf a0 = load_a_frag(Az, lda, row0,      k0, lane);
        v16bf a1 = load_a_frag(Az, lda, row0 + 16, k0, lane);
#pragma unroll
        for (int j = 0; j < 4; ++j) {
            v16bf b = load_bt_frag(Bz, ldbt, k0, col0 + j * 16, lane);
            acc[0][j] = __builtin_amdgcn_wmma_f32_16x16x32_bf16(
                false, a0, false, b, (short)0, acc[0][j], false, false);
            acc[1][j] = __builtin_amdgcn_wmma_f32_16x16x32_bf16(
                false, a1, false, b, (short)0, acc[1][j], false, false);
        }
    }

#pragma unroll
    for (int mi = 0; mi < 2; ++mi)
#pragma unroll
        for (int j = 0; j < 4; ++j) {
            const int cn = col0 + j * 16 + (lane & 15);
#pragma unroll
            for (int r = 0; r < 8; ++r) {
                const int rn = row0 + mi * 16 + r + (lane >> 4) * 8;
                const float v = acc[mi][j][r];
                if (EPI == 0) {
                    ((float*)C + (size_t)z * cStr)[(size_t)rn * ldc + cn] = v;
                } else if (EPI == 1) {
                    ((float*)C + (size_t)z * cStr)[(size_t)rn * ldc + cn] = fmaxf(v, 0.0f);
                } else if (EPI == 2) {
                    ((bhalf*)C + (size_t)z * cStr)[(size_t)rn * ldc + cn] =
                        (bhalf)((cn < rn) ? v : 0.0f);
                } else {
                    const float g = gate[(size_t)z * gStr + (size_t)rn * ldg + cn];
                    ((bhalf*)C + (size_t)z * cStr)[(size_t)rn * ldc + cn] =
                        (bhalf)(fmaxf(v, 0.0f) * g);
                }
            }
        }
}

// ---------------- fp32 -> bf16 tiled transpose (LDS, one-shot for weights) --
// src: [R x C] fp32 row-major (+batch), dst: [C x R] bf16 row-major (+batch)

__global__ __launch_bounds__(256) void transpose_cvt_kernel(
    const float* __restrict__ src, bhalf* __restrict__ dst,
    int R, int C, long long sStr, long long dStr) {
    __shared__ float tile[32][33];   // +1 pad: conflict-free transpose
    const float* s = src + (size_t)blockIdx.z * sStr;
    bhalf*       d = dst + (size_t)blockIdx.z * dStr;
    const int r0 = blockIdx.y * 32, c0 = blockIdx.x * 32;
#pragma unroll
    for (int i = 0; i < 4; ++i) {
        const int idx = threadIdx.x + i * 256;
        tile[idx >> 5][idx & 31] = s[(size_t)(r0 + (idx >> 5)) * C + (c0 + (idx & 31))];
    }
    __syncthreads();
#pragma unroll
    for (int i = 0; i < 4; ++i) {
        const int idx = threadIdx.x + i * 256;
        const int cc = idx >> 5, rr = idx & 31;
        d[(size_t)(c0 + cc) * R + (r0 + rr)] = (bhalf)tile[rr][cc];
    }
}

// ---------------- TDM warm: tensor_load_to_lds + s_wait_tensorcnt -----------
// Pulls the head of the decoder weights toward GL2 via the Tensor Data Mover
// before the layer loop starts. Descriptor packing is approximate (fields per
// CDNA5 ISA); the DMA result is only used to warm
// caches, never read.

__global__ __launch_bounds__(32) void tdm_l2_warm_kernel(const void* gsrc) {
#if defined(__has_builtin) && __has_builtin(__builtin_amdgcn_tensor_load_to_lds)
    __shared__ unsigned char stage[16384];
    if (threadIdx.x == 0) {
        const unsigned long long ga = (unsigned long long)(size_t)gsrc;
        const unsigned lds_off = (unsigned)(size_t)(void*)&stage[0];
        u32x4 g0 = { 1u,                                   // count=1 (user D#)
                     lds_off,                              // lds_addr
                     (unsigned)ga,                         // global_addr[31:0]
                     (unsigned)((ga >> 32) & 0x01ffffffu)
                         | (2u << 30) };                   // addr hi | type=2
        i32x8 g1 = { (int)(2u << 16),                      // data_size=4B, mask=0
                     0,                                    // no atomic barrier
                     4096,                                 // tensor_dim0 (approx)
                     1,                                    // tensor_dim1 (approx)
                     4096 | (1 << 16),                     // tile_dim0/1 (approx)
                     0, 0, 0 };
        i32x4 g2 = { 0, 0, 0, 0 };
        i32x4 g3 = { 0, 0, 0, 0 };
#if __clang_major__ >= 23
        i32x8 g4 = { 0, 0, 0, 0, 0, 0, 0, 0 };
        __builtin_amdgcn_tensor_load_to_lds(g0, g1, g2, g3, g4, 0);
#else
        __builtin_amdgcn_tensor_load_to_lds(g0, g1, g2, g3, 0);
#endif
        __builtin_amdgcn_s_wait_tensorcnt(0);
    }
#else
    (void)gsrc;
#endif
}

// ---------------- RoPE: x_sp (fp32) -> qr (bf16) ---------------------------
// pair p (n=2p,2p+1): freq = THETA^(-2p/N)/(2pi) = 2^(-p/256)/(2pi)

__global__ __launch_bounds__(256) void rope_kernel(const float* __restrict__ xsp,
                                                   bhalf* __restrict__ qr) {
    const size_t i = (size_t)blockIdx.x * blockDim.x + threadIdx.x;  // pair idx
    const int p = (int)(i % (N_LAT / 2));
    const int t = (int)((i / (N_LAT / 2)) % T_SEQ);
    const float v0 = xsp[2 * i];
    const float v1 = xsp[2 * i + 1];
    float phase = (float)t * (exp2f(-(float)p * (1.0f / 256.0f)) * INV_2PI);
    phase -= floorf(phase);
    const float ph = phase * TWO_PI;
    const float c = __cosf(ph), s = __sinf(ph);
    qr[2 * i]     = (bhalf)(v0 * c - v1 * s);
    qr[2 * i + 1] = (bhalf)(v1 * c + v0 * s);
}

// ---------------- LayerNorm helpers (one wave32 per 256-wide row) ----------

__device__ __forceinline__ float wsum(float v) {
#pragma unroll
    for (int m = 16; m > 0; m >>= 1) v += __shfl_xor(v, m, 32);
    return v;
}

// x = LN(embed[idx]); writes x (f32), xb (bf16 row-major), xT (bf16 [D,T])
__global__ __launch_bounds__(256) void embed_ln_kernel(
    const int* __restrict__ idx, const float* __restrict__ embed,
    float* __restrict__ x, bhalf* __restrict__ xb, bhalf* __restrict__ xT) {
    const int lane = threadIdx.x & 31;
    const size_t row = (size_t)blockIdx.x * 8 + (threadIdx.x >> 5);
    const float* p = embed + (size_t)idx[row] * D_EMB;
    float v[8], s = 0.f;
#pragma unroll
    for (int e = 0; e < 8; ++e) { v[e] = p[e * 32 + lane]; s += v[e]; }
    const float mean = wsum(s) * (1.0f / D_EMB);
    float q = 0.f;
#pragma unroll
    for (int e = 0; e < 8; ++e) { const float d = v[e] - mean; q += d * d; }
    const float inv = rsqrtf(wsum(q) * (1.0f / D_EMB) + LN_EPS);
#pragma unroll
    for (int e = 0; e < 8; ++e) {
        const int col = e * 32 + lane;
        const float o = (v[e] - mean) * inv;
        x [row * D_EMB + col] = o;
        xb[row * D_EMB + col] = (bhalf)o;
        xT[(size_t)col * T_SEQ + row] = (bhalf)o;
    }
}

__global__ __launch_bounds__(256) void ln_bf16_kernel(
    const float* __restrict__ src, bhalf* __restrict__ dst) {
    const int lane = threadIdx.x & 31;
    const size_t row = (size_t)blockIdx.x * 8 + (threadIdx.x >> 5);
    const float* p = src + row * D_EMB;
    float v[8], s = 0.f;
#pragma unroll
    for (int e = 0; e < 8; ++e) { v[e] = p[e * 32 + lane]; s += v[e]; }
    const float mean = wsum(s) * (1.0f / D_EMB);
    float q = 0.f;
#pragma unroll
    for (int e = 0; e < 8; ++e) { const float d = v[e] - mean; q += d * d; }
    const float inv = rsqrtf(wsum(q) * (1.0f / D_EMB) + LN_EPS);
#pragma unroll
    for (int e = 0; e < 8; ++e)
        dst[row * D_EMB + e * 32 + lane] = (bhalf)((v[e] - mean) * inv);
}

// x = LN(x + LN(y)); writes x (f32), xb (bf16), xT (bf16 [D,T])
__global__ __launch_bounds__(256) void ln_residual_kernel(
    const float* __restrict__ y, float* __restrict__ x,
    bhalf* __restrict__ xb, bhalf* __restrict__ xT) {
    const int lane = threadIdx.x & 31;
    const size_t row = (size_t)blockIdx.x * 8 + (threadIdx.x >> 5);
    float v[8], s = 0.f;
#pragma unroll
    for (int e = 0; e < 8; ++e) { v[e] = y[row * D_EMB + e * 32 + lane]; s += v[e]; }
    float mean = wsum(s) * (1.0f / D_EMB);
    float q = 0.f;
#pragma unroll
    for (int e = 0; e < 8; ++e) { const float d = v[e] - mean; q += d * d; }
    float inv = rsqrtf(wsum(q) * (1.0f / D_EMB) + LN_EPS);
    float u[8]; s = 0.f;
#pragma unroll
    for (int e = 0; e < 8; ++e) {
        u[e] = x[row * D_EMB + e * 32 + lane] + (v[e] - mean) * inv;
        s += u[e];
    }
    mean = wsum(s) * (1.0f / D_EMB);
    q = 0.f;
#pragma unroll
    for (int e = 0; e < 8; ++e) { const float d = u[e] - mean; q += d * d; }
    inv = rsqrtf(wsum(q) * (1.0f / D_EMB) + LN_EPS);
#pragma unroll
    for (int e = 0; e < 8; ++e) {
        const int col = e * 32 + lane;
        const float o = (u[e] - mean) * inv;
        x [row * D_EMB + col] = o;
        xb[row * D_EMB + col] = (bhalf)o;
        xT[(size_t)col * T_SEQ + row] = (bhalf)o;
    }
}

// ---------------- driver ---------------------------------------------------

extern "C" void kernel_launch(void* const* d_in, const int* in_sizes, int n_in,
                              void* d_out, int out_size, void* d_ws, size_t ws_size,
                              hipStream_t stream) {
    (void)in_sizes; (void)n_in; (void)out_size; (void)ws_size;
    const int*   idx      = (const int*)  d_in[0];
    const float* embed    = (const float*)d_in[1];
    const float* encoder  = (const float*)d_in[2];   // [NH, D, N]
    const float* encoderv = (const float*)d_in[3];   // [NH, D, N]
    const float* decoder  = (const float*)d_in[4];   // [HN, D]
    const float* lm_head  = (const float*)d_in[5];   // [D, VOCAB]
    float*       out      = (float*)d_out;           // [T, VOCAB]

    char* wsp = (char*)d_ws;
    auto give = [&](size_t bytes) -> void* {
        void* p = (void*)wsp;
        wsp += (bytes + 255) & ~(size_t)255;
        return p;
    };
    float* x    = (float*)give((size_t)T_SEQ * D_EMB * 4);
    bhalf* xb   = (bhalf*)give((size_t)T_SEQ * D_EMB * 2);
    bhalf* xT   = (bhalf*)give((size_t)D_EMB * T_SEQ * 2);           // x^T
    float* y5   = (float*)give((size_t)T_SEQ * D_EMB * 4);
    float* yKV  = (float*)give((size_t)NHEAD * T_SEQ * D_EMB * 4);
    bhalf* yKVb = (bhalf*)give((size_t)NHEAD * T_SEQ * D_EMB * 2);
    float* xsp  = (float*)give((size_t)NHEAD * T_SEQ * N_LAT * 4);   // 128 MB
    bhalf* qr   = (bhalf*)give((size_t)NHEAD * T_SEQ * N_LAT * 2);   //  64 MB
    bhalf* sc   = (bhalf*)give((size_t)NHEAD * T_SEQ * T_SEQ * 2);   //   8 MB
    bhalf* xy   = (bhalf*)give((size_t)T_SEQ * HN * 2);              //  64 MB
    bhalf* encT = (bhalf*)give((size_t)NHEAD * N_LAT * D_EMB * 2);   //  16 MB [NH,N,D]
    bhalf* encvT= (bhalf*)give((size_t)NHEAD * N_LAT * D_EMB * 2);   //  16 MB
    bhalf* decT = (bhalf*)give((size_t)D_EMB * HN * 2);              //  16 MB [D,HN]
    bhalf* lmT  = (bhalf*)give((size_t)VOCAB * D_EMB * 2);           //  [VOCAB,D]

    const dim3 blk(256);

    // One-shot weight convert+transpose to bf16 (B-side contiguous layout).
    transpose_cvt_kernel<<<dim3(N_LAT / 32, D_EMB / 32, NHEAD), blk, 0, stream>>>(
        encoder, encT, D_EMB, N_LAT, (long long)D_EMB * N_LAT, (long long)N_LAT * D_EMB);
    transpose_cvt_kernel<<<dim3(N_LAT / 32, D_EMB / 32, NHEAD), blk, 0, stream>>>(
        encoderv, encvT, D_EMB, N_LAT, (long long)D_EMB * N_LAT, (long long)N_LAT * D_EMB);
    transpose_cvt_kernel<<<dim3(D_EMB / 32, HN / 32, 1), blk, 0, stream>>>(
        decoder, decT, HN, D_EMB, 0, 0);
    transpose_cvt_kernel<<<dim3(VOCAB / 32, D_EMB / 32, 1), blk, 0, stream>>>(
        lm_head, lmT, D_EMB, VOCAB, 0, 0);

    // TDM warm of decoder weights toward GL2 (async-tensor path).
    tdm_l2_warm_kernel<<<1, 32, 0, stream>>>(decoder);

    embed_ln_kernel<<<T_SEQ / 8, blk, 0, stream>>>(idx, embed, x, xb, xT);

    for (int layer = 0; layer < N_LAYER; ++layer) {
        // x_sp[h] = relu(x @ encoder[h])          [T x N], heads in grid.z
        gemm_wmma<1><<<dim3(N_LAT / 256, T_SEQ / 64, NHEAD), blk, 0, stream>>>(
            xb, D_EMB, 0,
            encT, D_EMB, (long long)N_LAT * D_EMB,
            xsp, N_LAT, (long long)T_SEQ * N_LAT,
            nullptr, 0, 0, D_EMB);

        // qr = rope(x_sp) in bf16
        rope_kernel<<<(unsigned)((size_t)NHEAD * T_SEQ * (N_LAT / 2) / 256),
                      blk, 0, stream>>>(xsp, qr);

        // scores[h] = strict_lower_mask(qr qr^T)  [T x T] -> bf16
        gemm_wmma<2><<<dim3(T_SEQ / 256, T_SEQ / 64, NHEAD), blk, 0, stream>>>(
            qr, N_LAT, (long long)T_SEQ * N_LAT,
            qr, N_LAT, (long long)T_SEQ * N_LAT,
            sc, T_SEQ, (long long)T_SEQ * T_SEQ,
            nullptr, 0, 0, N_LAT);

        // yKV[h] = scores[h] @ x                  [T x D]  (B = x^T, contiguous)
        gemm_wmma<0><<<dim3(D_EMB / 256, T_SEQ / 64, NHEAD), blk, 0, stream>>>(
            sc, T_SEQ, (long long)T_SEQ * T_SEQ,
            xT, T_SEQ, 0,
            yKV, D_EMB, (long long)T_SEQ * D_EMB,
            nullptr, 0, 0, T_SEQ);

        // yKVb = LN(yKV) -> bf16
        ln_bf16_kernel<<<(NHEAD * T_SEQ) / 8, blk, 0, stream>>>(yKV, yKVb);

        // xy[:, h*N : (h+1)*N] = relu(yKVb[h] @ encoder_v[h]) * x_sp[h] -> bf16
        gemm_wmma<3><<<dim3(N_LAT / 256, T_SEQ / 64, NHEAD), blk, 0, stream>>>(
            yKVb, D_EMB, (long long)T_SEQ * D_EMB,
            encvT, D_EMB, (long long)N_LAT * D_EMB,
            xy, HN, (long long)N_LAT /* column offset per head */,
            xsp, N_LAT, (long long)T_SEQ * N_LAT, D_EMB);

        // y5 = xy @ decoder                       [T x D]  (B = decT [D,HN])
        gemm_wmma<0><<<dim3(D_EMB / 256, T_SEQ / 64, 1), blk, 0, stream>>>(
            xy, HN, 0,
            decT, HN, 0,
            y5, D_EMB, 0,
            nullptr, 0, 0, HN);

        // x = LN(x + LN(y5))
        ln_residual_kernel<<<T_SEQ / 8, blk, 0, stream>>>(y5, x, xb, xT);
    }

    // logits = x @ lm_head                        [T x VOCAB] -> d_out
    gemm_wmma<0><<<dim3(VOCAB / 256, T_SEQ / 64, 1), blk, 0, stream>>>(
        xb, D_EMB, 0,
        lmT, D_EMB, 0,
        out, VOCAB, 0,
        nullptr, 0, 0, D_EMB);
}
